// DWT_MSAAB_IDWT_Block_70334384439938
// MI455X (gfx1250) — matrix operand order repaired and checked by reading, not compiled, not run on hardware
//
#include <hip/hip_runtime.h>
#include <hip/hip_bf16.h>

// ---------------------------------------------------------------------------
// Types for WMMA fragments (gfx1250, wave32)
// ---------------------------------------------------------------------------
typedef __attribute__((ext_vector_type(16))) _Float16 v16h;
typedef __attribute__((ext_vector_type(8)))  float    v8f;
typedef __attribute__((ext_vector_type(4)))  unsigned int v4u;
typedef __attribute__((ext_vector_type(4)))  float    v4f;
typedef __attribute__((ext_vector_type(4)))  int      v4i;

union AF { v16h h; v4u u[2]; };
union H8 { v4u u; _Float16 h[8]; };
union H4 { unsigned long long q; _Float16 h[4]; };

// CDNA5 async global->LDS staging (guarded; falls back to plain loads)
#if __has_builtin(__builtin_amdgcn_global_load_async_to_lds_b128) && \
    __has_builtin(__builtin_amdgcn_s_wait_asynccnt)
#define HAVE_ASYNC_LDS 1
typedef __attribute__((address_space(1))) v4i* gv4ip;   // global int4*
typedef __attribute__((address_space(3))) v4i* lv4ip;   // LDS int4*
#else
#define HAVE_ASYNC_LDS 0
#endif

#define LEAK   0.3f
#define BN_EPS 0.001f

// Problem dims (fixed by the reference)
#define BATCH 8
#define HS 128              // half-res H
#define WSP 128             // half-res W
#define NP (BATCH*HS*WSP)   // 131072 half-res pixels
#define FH 256
#define FW 256
#define CF 64

enum { EPI_LRELU = 0, EPI_SIGMUL = 1, EPI_BNSILU = 2, EPI_LIN32 = 3 };

// ---------------------------------------------------------------------------
// Weight convert + transpose:  w[tap][cin][cout] (f32) -> wT[tap][cout][cin] (f16)
// ---------------------------------------------------------------------------
__global__ void k_wcvt(const float* __restrict__ w, _Float16* __restrict__ wT,
                       int taps, int Cin, int Cout)
{
    int t = blockIdx.x * blockDim.x + threadIdx.x;
    int total = taps * Cin * Cout;
    if (t >= total) return;
    int co  = t % Cout;
    int ci  = (t / Cout) % Cin;
    int tap = t / (Cout * Cin);
    wT[((size_t)tap * Cout + co) * Cin + ci] = (_Float16)w[t];
}

// ---------------------------------------------------------------------------
// Haar DWT (4 channels/thread)
// ---------------------------------------------------------------------------
__global__ void k_dwt(const float* __restrict__ X, float* __restrict__ hf32,
                      _Float16* __restrict__ hfh, _Float16* __restrict__ LLh)
{
    int t = blockIdx.x * blockDim.x + threadIdx.x;
    if (t >= NP * (CF / 4)) return;
    int c = (t % (CF / 4)) * 4;
    int p = t / (CF / 4);
    int x = p % WSP;
    int y = (p / WSP) % HS;
    int b = p / (WSP * HS);
    size_t base = (((size_t)b * FH + 2 * y) * FW + 2 * x) * CF + c;
    v4f a  = *(const v4f*)(X + base);
    v4f bb = *(const v4f*)(X + base + CF);
    v4f cc = *(const v4f*)(X + base + (size_t)FW * CF);
    v4f dd = *(const v4f*)(X + base + (size_t)FW * CF + CF);
    v4f LL = (a + bb + cc + dd) * 0.5f;
    v4f LH = (cc + dd - a - bb) * 0.5f;
    v4f HL = (bb + dd - a - cc) * 0.5f;
    v4f HH = (a - bb - cc + dd) * 0.5f;
    size_t hp = (size_t)p * 192;
    *(v4f*)(hf32 + hp + c)       = LH;
    *(v4f*)(hf32 + hp + 64 + c)  = HL;
    *(v4f*)(hf32 + hp + 128 + c) = HH;
    H4 h;
    #pragma unroll
    for (int i = 0; i < 4; ++i) h.h[i] = (_Float16)LH[i];
    *(unsigned long long*)(hfh + hp + c) = h.q;
    #pragma unroll
    for (int i = 0; i < 4; ++i) h.h[i] = (_Float16)HL[i];
    *(unsigned long long*)(hfh + hp + 64 + c) = h.q;
    #pragma unroll
    for (int i = 0; i < 4; ++i) h.h[i] = (_Float16)HH[i];
    *(unsigned long long*)(hfh + hp + 128 + c) = h.q;
    #pragma unroll
    for (int i = 0; i < 4; ++i) h.h[i] = (_Float16)LL[i];
    *(unsigned long long*)(LLh + (size_t)p * CF + c) = h.q;
}

// ---------------------------------------------------------------------------
// Implicit-GEMM conv via v_wmma_f32_16x16x32_f16, compile-time shapes.
//   block = 256 threads = 8 waves; wave owns MT 16-pixel M tiles x COUT.
//   MT=2 reuses each B fragment across 2 WMMAs (halves LDS B-traffic/op).
//   Per-tap weight slab staged to LDS (async global->LDS when available).
// ---------------------------------------------------------------------------
template<int KH, int KW, int DIL, int CIN, int COUT, int MT, int EPI>
__global__ __launch_bounds__(256)
void conv_wmma(const _Float16* __restrict__ in, const _Float16* __restrict__ wT,
               const float* __restrict__ bias,
               _Float16* __restrict__ outh, float* __restrict__ outf,
               const float* __restrict__ mulsrc,
               const float* __restrict__ bng, const float* __restrict__ bnb,
               const float* __restrict__ bnm, const float* __restrict__ bnv)
{
    constexpr int NB = COUT / 16;          // N tiles per wave
    __shared__ __align__(16) _Float16 wlds[COUT * CIN];   // current tap slab
    __shared__ __align__(16) _Float16 stg[8][16 * 16];    // per-wave D staging

    const int tid  = threadIdx.x;
    const int lane = tid & 31;
    const int wave = tid >> 5;
    const int m0   = blockIdx.x * (128 * MT) + wave * (16 * MT);
    const int l15  = lane & 15;
    const bool hi  = lane >= 16;
    const int xb   = m0 % WSP;             // m0 multiple of 16*MT -> row aligned
    const int y    = (m0 / WSP) % HS;
    const int bidx = m0 / (WSP * HS);
    const int kA   = hi ? 8 : 0;           // A-fragment K base offset
    const int kB   = hi ? 8 : 0;           // B-fragment K base offset

    v8f acc[MT][NB];
    #pragma unroll
    for (int mt = 0; mt < MT; ++mt)
        #pragma unroll
        for (int nb = 0; nb < NB; ++nb) { v8f z = {}; acc[mt][nb] = z; }

    for (int tap = 0; tap < KH * KW; ++tap) {
        // --- stage this tap's weights into LDS (all 256 threads) ---
        __syncthreads();
        {
            const _Float16* src = wT + (size_t)tap * COUT * CIN;
            constexpr int CHUNKS = COUT * CIN / 8;
            #pragma unroll
            for (int i = 0; i < (CHUNKS + 255) / 256; ++i) {
                int idx = tid + i * 256;
                if (CHUNKS % 256 == 0 || idx < CHUNKS) {
#if HAVE_ASYNC_LDS
                    __builtin_amdgcn_global_load_async_to_lds_b128(
                        (gv4ip)(src + idx * 8),
                        (lv4ip)(wlds + idx * 8), 0, 0);
#else
                    ((v4u*)wlds)[idx] = ((const v4u*)src)[idx];
#endif
                }
            }
#if HAVE_ASYNC_LDS
            __builtin_amdgcn_s_wait_asynccnt(0);
#endif
        }
        __syncthreads();

        const int kh = tap / KW, kw = tap % KW;
        const int dy = kh * DIL - ((KH - 1) * DIL) / 2;
        const int dx = kw * DIL - ((KW - 1) * DIL) / 2;
        const int iy = y + dy;
        const bool vy = (iy >= 0) & (iy < HS);
        const _Float16* arow0 =
            in + (long long)((bidx * HS + iy) * WSP) * CIN;

        bool val[MT];
        const _Float16* arow[MT];
        #pragma unroll
        for (int mt = 0; mt < MT; ++mt) {
            const int ix = xb + mt * 16 + l15 + dx;
            val[mt]  = vy & (ix >= 0) & (ix < WSP);
            arow[mt] = arow0 + (long long)ix * CIN;
            if (val[mt]) __builtin_prefetch(arow[mt], 0, 1); // global_prefetch_b8
        }

        #pragma unroll
        for (int kc = 0; kc < CIN; kc += 32) {
            AF a[MT];
            #pragma unroll
            for (int mt = 0; mt < MT; ++mt) {
                if (val[mt]) {
                    const v4u* ap = (const v4u*)(arow[mt] + kc + kA);
                    a[mt].u[0] = ap[0];          // K = kA .. kA+7
                    a[mt].u[1] = ap[2];          // K = kA+16 .. kA+23
                } else {
                    v4u z = {}; a[mt].u[0] = z; a[mt].u[1] = z;
                }
            }
            #pragma unroll
            for (int nb = 0; nb < NB; ++nb) {
                const int n = (nb << 4) + l15;
                const v4u* bp = (const v4u*)(wlds + n * CIN + kc + kB);
                AF bfr; bfr.u[0] = bp[0]; bfr.u[1] = bp[2];
                #pragma unroll
                for (int mt = 0; mt < MT; ++mt) {
                    acc[mt][nb] = __builtin_amdgcn_wmma_f32_16x16x32_f16(
                        false, a[mt].h, false, bfr.h, (short)0,
                        acc[mt][nb], false, false);
                }
            }
        }
    }

    // --- epilogue ---
    _Float16* my = &stg[wave][0];
    #pragma unroll
    for (int mt = 0; mt < MT; ++mt) {
        const int mbase = m0 + mt * 16;
        #pragma unroll
        for (int nb = 0; nb < NB; ++nb) {
            const int n = (nb << 4) + l15;
            const float bs = bias[n];
            if constexpr (EPI == EPI_LRELU || EPI == EPI_SIGMUL) {
                // lane holds column n, rows i+(hi?8:0); transpose via wave LDS
                #pragma unroll
                for (int i = 0; i < 8; ++i) {
                    const int row = i + (hi ? 8 : 0);
                    float v = acc[mt][nb][i] + bs;
                    _Float16 hv;
                    if constexpr (EPI == EPI_LRELU) {
                        v = (v >= 0.f) ? v : LEAK * v;
                        hv = (_Float16)v;
                    } else {
                        float s = 1.f / (1.f + __expf(-v));
                        hv = (_Float16)(s * mulsrc[(size_t)(mbase + row) * COUT + n]);
                    }
                    my[row * 16 + l15] = hv;               // ds_store_b16
                }
                // same-wave LDS ops complete in order: safe to read back
                const int row = l15;
                const int ch  = hi ? 8 : 0;
                v4u tdat = *(const v4u*)(my + row * 16 + ch);   // ds_load_b128
                *(v4u*)(outh + (size_t)(mbase + row) * COUT + (nb << 4) + ch) = tdat;
            } else {
                #pragma unroll
                for (int i = 0; i < 8; ++i) {
                    const int pix = mbase + i + (hi ? 8 : 0);
                    float v = acc[mt][nb][i] + bs;
                    if constexpr (EPI == EPI_BNSILU) {
                        float z = (v - bnm[n]) * rsqrtf(bnv[n] + BN_EPS) * bng[n] + bnb[n];
                        outf[(size_t)pix * COUT + n] = z * (1.f / (1.f + __expf(-z)));
                    } else {
                        outf[(size_t)pix * COUT + n] = v;
                    }
                }
            }
        }
    }
}

// ---------------------------------------------------------------------------
// added = x1 + x2 + x3   (8 halves / thread)
// ---------------------------------------------------------------------------
__global__ void k_add3(const _Float16* __restrict__ a, const _Float16* __restrict__ b,
                       const _Float16* __restrict__ c, _Float16* __restrict__ o)
{
    int t = blockIdx.x * blockDim.x + threadIdx.x;
    if (t >= NP * CF / 8) return;
    size_t base = (size_t)t * 8;
    H8 A, B, C, O;
    A.u = *(const v4u*)(a + base);
    B.u = *(const v4u*)(b + base);
    C.u = *(const v4u*)(c + base);
    #pragma unroll
    for (int i = 0; i < 8; ++i)
        O.h[i] = (_Float16)((float)A.h[i] + (float)B.h[i] + (float)C.h[i]);
    *(v4u*)(o + base) = O.u;
}

// ---------------------------------------------------------------------------
// ll_exp = concat([LL, phf], channel) -> (NP, 256) f16   (8 halves / thread)
// ---------------------------------------------------------------------------
__global__ void k_llexp(const _Float16* __restrict__ LLh,
                        const _Float16* __restrict__ phfh, _Float16* __restrict__ o)
{
    int t = blockIdx.x * blockDim.x + threadIdx.x;
    if (t >= NP * 256 / 8) return;
    int ch = (t % 32) * 8;
    int p  = t / 32;
    v4u d = (ch < 64) ? *(const v4u*)(LLh + (size_t)p * 64 + ch)
                      : *(const v4u*)(phfh + (size_t)p * 192 + (ch - 64));
    *(v4u*)(o + (size_t)p * 256 + ch) = d;
}

// ---------------------------------------------------------------------------
// Inverse Haar DWT -> out(8,256,256,64) f32   (4 channels / thread)
// ---------------------------------------------------------------------------
__global__ void k_idwt(const float* __restrict__ pLL, const float* __restrict__ phf2,
                       const float* __restrict__ hf32, float* __restrict__ out)
{
    int t = blockIdx.x * blockDim.x + threadIdx.x;
    if (t >= BATCH * FH * FW * (CF / 4)) return;
    int c = (t % (CF / 4)) * 4;
    int x = (t / (CF / 4)) % FW;
    int y = (t / ((CF / 4) * FW)) % FH;
    int b = t / ((CF / 4) * FW * FH);
    int hx = x >> 1, hy = y >> 1, sx = x & 1, sy = y & 1;
    size_t p = ((size_t)b * HS + hy) * WSP + hx;
    v4f LL = *(const v4f*)(pLL + p * 64 + c);
    v4f LH = *(const v4f*)(phf2 + p * 64 + c);
    v4f HL = *(const v4f*)(hf32 + p * 192 + 64 + c);
    v4f HH = *(const v4f*)(hf32 + p * 192 + 128 + c);
    v4f v;
    if (sy == 0 && sx == 0)      v = (LL - LH - HL + HH) * 0.5f;
    else if (sy == 0)            v = (LL - LH + HL - HH) * 0.5f;
    else if (sx == 0)            v = (LL + LH - HL - HH) * 0.5f;
    else                         v = (LL + LH + HL + HH) * 0.5f;
    *(v4f*)(out + (size_t)t * 4) = v;
}

// ---------------------------------------------------------------------------
// Host launcher
// ---------------------------------------------------------------------------
extern "C" void kernel_launch(void* const* d_in, const int* in_sizes, int n_in,
                              void* d_out, int out_size, void* d_ws, size_t ws_size,
                              hipStream_t stream)
{
    const float* x    = (const float*)d_in[0];
    const float* w15  = (const float*)d_in[1];  const float* b15 = (const float*)d_in[2];
    const float* w25  = (const float*)d_in[3];  const float* b25 = (const float*)d_in[4];
    const float* w13  = (const float*)d_in[5];  const float* b13 = (const float*)d_in[6];
    const float* w23  = (const float*)d_in[7];  const float* b23 = (const float*)d_in[8];
    const float* w12  = (const float*)d_in[9];  const float* b12 = (const float*)d_in[10];
    const float* w22  = (const float*)d_in[11]; const float* b22 = (const float*)d_in[12];
    const float* wd1  = (const float*)d_in[13]; const float* bd1 = (const float*)d_in[14];
    const float* wd2  = (const float*)d_in[15]; const float* bd2 = (const float*)d_in[16];
    const float* wd3  = (const float*)d_in[17]; const float* bd3 = (const float*)d_in[18];
    const float* wca  = (const float*)d_in[19]; const float* bca = (const float*)d_in[20];
    const float* wcb  = (const float*)d_in[21]; const float* bcb = (const float*)d_in[22];
    const float* bng  = (const float*)d_in[23]; const float* bnb = (const float*)d_in[24];
    const float* bnm  = (const float*)d_in[25]; const float* bnv = (const float*)d_in[26];
    const float* whf  = (const float*)d_in[27]; const float* bhf = (const float*)d_in[28];

    char* ws = (char*)d_ws;
    size_t off = 0;
    auto alloc = [&](size_t bytes) -> void* {
        void* p = ws + off;
        off = (off + bytes + 255) & ~(size_t)255;
        return p;
    };

    float*     hf32  = (float*)    alloc((size_t)NP * 192 * 4);
    _Float16*  hfh   = (_Float16*) alloc((size_t)NP * 192 * 2);
    _Float16*  LLh   = (_Float16*) alloc((size_t)NP * 64 * 2);
    _Float16*  tA    = (_Float16*) alloc((size_t)NP * 64 * 2);
    _Float16*  tB    = (_Float16*) alloc((size_t)NP * 64 * 2);
    _Float16*  x1h   = (_Float16*) alloc((size_t)NP * 64 * 2);
    _Float16*  x2h   = (_Float16*) alloc((size_t)NP * 64 * 2);
    _Float16*  x3h   = (_Float16*) alloc((size_t)NP * 64 * 2);
    _Float16*  addh  = (_Float16*) alloc((size_t)NP * 64 * 2);
    _Float16*  phfh  = (_Float16*) alloc((size_t)NP * 192 * 2);
    _Float16*  llex  = (_Float16*) alloc((size_t)NP * 256 * 2);
    float*     pLL   = (float*)    alloc((size_t)NP * 64 * 4);
    float*     phf2  = (float*)    alloc((size_t)NP * 64 * 4);

    _Float16* w15T = (_Float16*)alloc((size_t)25 * 192 * 64 * 2);
    _Float16* w25T = (_Float16*)alloc((size_t)25 * 64 * 64 * 2);
    _Float16* w13T = (_Float16*)alloc((size_t)9 * 192 * 64 * 2);
    _Float16* w23T = (_Float16*)alloc((size_t)9 * 64 * 64 * 2);
    _Float16* w12T = (_Float16*)alloc((size_t)4 * 192 * 64 * 2);
    _Float16* w22T = (_Float16*)alloc((size_t)4 * 64 * 64 * 2);
    _Float16* wd1T = (_Float16*)alloc((size_t)9 * 64 * 64 * 2);
    _Float16* wd2T = (_Float16*)alloc((size_t)9 * 64 * 64 * 2);
    _Float16* wd3T = (_Float16*)alloc((size_t)9 * 64 * 64 * 2);
    _Float16* wcaT = (_Float16*)alloc((size_t)1 * 64 * 192 * 2);
    _Float16* wcbT = (_Float16*)alloc((size_t)9 * 256 * 64 * 2);
    _Float16* whfT = (_Float16*)alloc((size_t)1 * 192 * 64 * 2);

    auto cvt = [&](const float* w, _Float16* wT, int taps, int ci, int co) {
        int total = taps * ci * co;
        k_wcvt<<<(total + 255) / 256, 256, 0, stream>>>(w, wT, taps, ci, co);
    };
    cvt(w15, w15T, 25, 192, 64);  cvt(w25, w25T, 25, 64, 64);
    cvt(w13, w13T,  9, 192, 64);  cvt(w23, w23T,  9, 64, 64);
    cvt(w12, w12T,  4, 192, 64);  cvt(w22, w22T,  4, 64, 64);
    cvt(wd1, wd1T,  9, 64, 64);   cvt(wd2, wd2T,  9, 64, 64);
    cvt(wd3, wd3T,  9, 64, 64);   cvt(wca, wcaT,  1, 64, 192);
    cvt(wcb, wcbT,  9, 256, 64);  cvt(whf, whfT,  1, 192, 64);

    k_dwt<<<(NP * (CF / 4) + 255) / 256, 256, 0, stream>>>(x, hf32, hfh, LLh);

    dim3 g2(NP / 256), g1(NP / 128), cblk(256);
    // branch 1 (5x5 -> 5x5)
    conv_wmma<5,5,1,192,64,2,EPI_LRELU><<<g2, cblk, 0, stream>>>(hfh, w15T, b15,
        tA, nullptr, nullptr, nullptr, nullptr, nullptr, nullptr);
    conv_wmma<5,5,1,64,64,2,EPI_LRELU><<<g2, cblk, 0, stream>>>(tA, w25T, b25,
        x1h, nullptr, nullptr, nullptr, nullptr, nullptr, nullptr);
    // branch 2 (3x3 -> 3x3)
    conv_wmma<3,3,1,192,64,2,EPI_LRELU><<<g2, cblk, 0, stream>>>(hfh, w13T, b13,
        tA, nullptr, nullptr, nullptr, nullptr, nullptr, nullptr);
    conv_wmma<3,3,1,64,64,2,EPI_LRELU><<<g2, cblk, 0, stream>>>(tA, w23T, b23,
        x2h, nullptr, nullptr, nullptr, nullptr, nullptr, nullptr);
    // branch 3 (2x2 -> 2x2)
    conv_wmma<2,2,1,192,64,2,EPI_LRELU><<<g2, cblk, 0, stream>>>(hfh, w12T, b12,
        tA, nullptr, nullptr, nullptr, nullptr, nullptr, nullptr);
    conv_wmma<2,2,1,64,64,2,EPI_LRELU><<<g2, cblk, 0, stream>>>(tA, w22T, b22,
        x3h, nullptr, nullptr, nullptr, nullptr, nullptr, nullptr);

    k_add3<<<(NP * CF / 8 + 255) / 256, 256, 0, stream>>>(x1h, x2h, x3h, addh);

    // dilated 3x3 chain
    conv_wmma<3,3,2,64,64,2,EPI_LRELU><<<g2, cblk, 0, stream>>>(addh, wd1T, bd1,
        tA, nullptr, nullptr, nullptr, nullptr, nullptr, nullptr);
    conv_wmma<3,3,2,64,64,2,EPI_LRELU><<<g2, cblk, 0, stream>>>(tA, wd2T, bd2,
        tB, nullptr, nullptr, nullptr, nullptr, nullptr, nullptr);
    conv_wmma<3,3,2,64,64,2,EPI_LRELU><<<g2, cblk, 0, stream>>>(tB, wd3T, bd3,
        tA, nullptr, nullptr, nullptr, nullptr, nullptr, nullptr);

    // 1x1 gate: phf = sigmoid(conv(y)) * hf   (fused epilogue; COUT=192 -> MT=1)
    conv_wmma<1,1,1,64,192,1,EPI_SIGMUL><<<g1, cblk, 0, stream>>>(tA, wcaT, bca,
        phfh, nullptr, hf32, nullptr, nullptr, nullptr, nullptr);

    k_llexp<<<(NP * 256 / 8 + 255) / 256, 256, 0, stream>>>(LLh, phfh, llex);

    // 3x3 conv on concat + BN + SiLU -> pLL (f32)
    conv_wmma<3,3,1,256,64,2,EPI_BNSILU><<<g2, cblk, 0, stream>>>(llex, wcbT, bcb,
        nullptr, pLL, nullptr, bng, bnb, bnm, bnv);

    // 1x1 conv: phf2 (f32)
    conv_wmma<1,1,1,192,64,2,EPI_LIN32><<<g2, cblk, 0, stream>>>(phfh, whfT, bhf,
        nullptr, phf2, nullptr, nullptr, nullptr, nullptr, nullptr);

    k_idwt<<<(BATCH * FH * FW * (CF / 4) + 255) / 256, 256, 0, stream>>>(
        pLL, phf2, hf32, (float*)d_out);
}
